// Seq2SeqBasic_867583394175
// MI455X (gfx1250) — matrix operand order repaired and checked by reading, not compiled, not run on hardware
//
#include <hip/hip_runtime.h>

// ---------------------------------------------------------------------------
// MI455X (gfx1250) persistent-WGP seq2seq GRU.
//   - bf16 WMMA (v_wmma_f32_16x16x32_bf16), fp32 accumulate/activations
//   - single persistent workgroup, all recurrent state in LDS
//   - weights pre-packed to WMMA-B fragment layout, [Wx;Wh] stacked on K
//   - 4 independent accumulator chains per wave sharing one A fragment;
//     single B base pointer + immediate offsets; K-loop unroll capped at 2
//     to stay inside the 256-VGPR window (Round-2 version spilled).
// ---------------------------------------------------------------------------

typedef __bf16 bf16;
typedef __attribute__((ext_vector_type(8)))  bf16  v8bf;
typedef __attribute__((ext_vector_type(16))) bf16  v16bf;
typedef __attribute__((ext_vector_type(8)))  float v8f;

#define NB      64      // batch
#define NN      256     // hidden / feature
#define TE      512
#define TD      512
#define NTH     512     // threads (16 wave32)

// LDS strides (elements) and byte offsets
#define AST 520         // sA row stride (bf16): [x(256) | h(256)] + pad
#define RST 264         // sRH row stride (bf16)
#define UST 260         // sU row stride (f32)
#define SA_OFF    0
#define SRH_OFF   66560      // 64*520*2
#define SU_OFF    100352     // +64*264*2
#define SH0_OFF   166912     // +64*260*4
#define SH1_OFF   199680     // +64*256*2
#define SLEN_OFF  232448     // +64*256*2
#define SMEM_BYTES 232704    // +64*4 (rounded to 16)

#define GATE_ELEMS (512*512)
#define CAND_ELEMS (512*256)
#define LAYER_ELEMS (GATE_ELEMS + CAND_ELEMS)
#define CSTRIDE 32768        // elements between the 4 grouped N-tiles (4*16*512)

__device__ __forceinline__ bf16 f2bf(float f) {
    unsigned u = __float_as_uint(f);
    unsigned r = (u + 0x7FFFu + ((u >> 16) & 1u)) >> 16;   // RNE
    unsigned short s = (unsigned short)r;
    bf16 o; __builtin_memcpy(&o, &s, 2); return o;
}
__device__ __forceinline__ float bf2f(bf16 h) {
    unsigned short s; __builtin_memcpy(&s, &h, 2);
    return __uint_as_float(((unsigned)s) << 16);
}
__device__ __forceinline__ v16bf cat16(v8bf lo, v8bf hi) {
    return __builtin_shufflevector(lo, hi, 0,1,2,3,4,5,6,7,8,9,10,11,12,13,14,15);
}
__device__ __forceinline__ float sigm(float x) { return 1.0f / (1.0f + __expf(-x)); }

// ---------------------------------------------------------------------------
// Weight pack: fp32 [Wx(256,N); Wh(256,N)] -> bf16 WMMA-B fragment layout.
// Tile (kt,nt) covers K=[kt*32,+32), N=[nt*16,+16); linear tile = nt*16+kt.
// In-tile slot: lane = (k_local>=16 ? 16:0) | n_local, j = k & 15.
// ---------------------------------------------------------------------------
__global__ __launch_bounds__(256) void pack_weights_kernel(
    const float* __restrict__ enc_Wxg, const float* __restrict__ enc_Whg,
    const float* __restrict__ enc_Wxc, const float* __restrict__ enc_Whc,
    const float* __restrict__ dec_Wxg, const float* __restrict__ dec_Whg,
    const float* __restrict__ dec_Wxc, const float* __restrict__ dec_Whc,
    bf16* __restrict__ wpk)
{
    long id = (long)blockIdx.x * blockDim.x + threadIdx.x;
    if (id >= 4L * LAYER_ELEMS) return;
    int layer = (int)(id / LAYER_ELEMS);         // 0:enc0 1:enc1 2:dec0 3:dec1
    int rem   = (int)(id - (long)layer * LAYER_ELEMS);
    bool cand = rem >= GATE_ELEMS;
    int e     = cand ? rem - GATE_ELEMS : rem;
    int tile  = e >> 9, idx = e & 511;
    int lane  = idx >> 4, j = idx & 15;
    int kt    = tile & 15, nt = tile >> 4;
    int k     = kt * 32 + (lane >> 4) * 16 + j;
    int n     = nt * 16 + (lane & 15);
    int net   = layer >> 1, l = layer & 1;
    int Ncols = cand ? 256 : 512;
    const float* Wx = cand ? (net ? dec_Wxc : enc_Wxc) : (net ? dec_Wxg : enc_Wxg);
    const float* Wh = cand ? (net ? dec_Whc : enc_Whc) : (net ? dec_Whg : enc_Whg);
    Wx += (long)l * 256 * Ncols;
    Wh += (long)l * 256 * Ncols;
    float v = (k < 256) ? Wx[(long)k * Ncols + n] : Wh[(long)(k - 256) * Ncols + n];
    wpk[id] = f2bf(v);
}

// One WMMA K-tile step for the 4-tile group: shared A, 4 independent accs.
// All B addresses are immediate offsets off a single base pointer.
__device__ __forceinline__ void mm4(
    v8f acc[4], const bf16* __restrict__ arow, const bf16* __restrict__ bq,
    int kt, int bkt)
{
    v8bf alo = *(const v8bf*)(arow + kt * 32);
    v8bf ahi = *(const v8bf*)(arow + kt * 32 + 8);
    v16bf a = cat16(alo, ahi);
    #pragma unroll
    for (int c = 0; c < 4; ++c) {
        const bf16* p = bq + c * CSTRIDE + bkt * 512;
        v8bf blo = *(const v8bf*)(p);
        v8bf bhi = *(const v8bf*)(p + 8);
        __builtin_prefetch(p + 512, 0, 1);      // next K-tile, L2-resident
        v16bf bb = cat16(blo, bhi);
        acc[c] = __builtin_amdgcn_wmma_f32_16x16x32_bf16(
                     false, a, false, bb, (short)0, acc[c], false, false);
    }
}

// ---------------------------------------------------------------------------
// One gate-GEMM pass: 4 N-tiles sharing one A fragment. RPASS: nt in [0,16)
// -> r columns (fuse sigma, r*h); else nt in [16,32) -> u columns -> sU.
// ---------------------------------------------------------------------------
template<bool RPASS>
__device__ __forceinline__ void gate_pass(
    const bf16* __restrict__ arow,
    const bf16* __restrict__ Bg, const float* __restrict__ bg,
    bf16* sRH, float* sU, const bf16* sH,
    int g, int mr, int mlo, int lane)
{
    const int ntb = g + (RPASS ? 0 : 16);
    v8f acc[4];
    #pragma unroll
    for (int c = 0; c < 4; ++c) {
        float bv = bg[(ntb + 4 * c) * 16 + mlo];
        #pragma unroll
        for (int i = 0; i < 8; ++i) acc[c][i] = bv;
    }
    const bf16* bq = Bg + ((long)(ntb * 16) << 9) + (lane << 4);
    #pragma unroll 2
    for (int kt = 0; kt < 16; ++kt) mm4(acc, arow, bq, kt, kt);

    #pragma unroll
    for (int c = 0; c < 4; ++c) {
        int n = (ntb + 4 * c) * 16 + mlo;        // gate column
        if (RPASS) {                             // r: fuse r*h -> sRH
            #pragma unroll
            for (int i = 0; i < 8; ++i) {
                float r  = sigm(acc[c][i]);
                float hv = bf2f(sH[(mr + i) * NN + n]);
                sRH[(mr + i) * RST + n] = f2bf(r * hv);
            }
        } else {                                 // u -> sU
            int jn = n - 256;
            #pragma unroll
            for (int i = 0; i < 8; ++i)
                sU[(mr + i) * UST + jn] = sigm(acc[c][i]);
        }
    }
}

// ---------------------------------------------------------------------------
// One GRU layer step on the persistent workgroup.
// ---------------------------------------------------------------------------
template<bool XG, bool MASK, bool WOUT>
__device__ __forceinline__ void gru_layer(
    const float* __restrict__ xg, long xStride,        // XG: global x rows
    const bf16*  __restrict__ xl,                      // !XG: LDS bf16 (sH0)
    bf16* sA, bf16* sRH, float* sU, bf16* sH,
    const int* sLen, int t,
    const bf16* __restrict__ Bg, const bf16* __restrict__ Bc,
    const float* __restrict__ bg, const float* __restrict__ bc,
    float* __restrict__ outp, long outStride, int tid)
{
    // ---- stage A = [x | h] ----
    for (int w = tid; w < NB * NN; w += NTH) {
        int b = w >> 8, j = w & 255;
        bf16 xv = XG ? f2bf(xg[(long)b * xStride + j]) : xl[w];
        sA[b * AST + j]       = xv;
        sA[b * AST + 256 + j] = sH[w];
    }
    __syncthreads();

    const int lane = tid & 31;
    const int wave = tid >> 5;
    const int mlo  = lane & 15;
    const int grp  = lane >> 4;          // K sub-window select for A/C frags
    const int mt   = wave & 3;           // M tile row-block for this wave
    const int g    = wave >> 2;          // N-tile group (nt === g mod 4)
    const int mr   = mt * 16 + grp * 8;  // C-fragment base row

    // ---- gate GEMM: [x;h] @ Bg + bg  (M=64,K=512,N=512) ----
    {
        const bf16* arow = sA + (mt * 16 + mlo) * AST + grp * 16;
        gate_pass<true >(arow, Bg, bg, sRH, sU, sH, g, mr, mlo, lane); // r cols
        gate_pass<false>(arow, Bg, bg, sRH, sU, sH, g, mr, mlo, lane); // u cols
    }
    __syncthreads();

    // ---- cand GEMM: [x ; r*h] @ Bc + bc (M=64,K=512,N=256), fused combine
    {
        const bf16* arowA = sA  + (mt * 16 + mlo) * AST + grp * 16;  // k 0..255
        const bf16* arowR = sRH + (mt * 16 + mlo) * RST + grp * 16;  // k 256..511
        v8f acc[4];
        #pragma unroll
        for (int c = 0; c < 4; ++c) {
            float bv = bc[(g + 4 * c) * 16 + mlo];
            #pragma unroll
            for (int i = 0; i < 8; ++i) acc[c][i] = bv;
        }
        const bf16* bq = Bc + ((long)(g * 16) << 9) + (lane << 4);
        #pragma unroll 2
        for (int kt = 0; kt < 8; ++kt) mm4(acc, arowA, bq, kt, kt);      // x half
        #pragma unroll 2
        for (int kt = 0; kt < 8; ++kt) mm4(acc, arowR, bq, kt, kt + 8);  // r*h half

        #pragma unroll
        for (int c = 0; c < 4; ++c) {
            int n = (g + 4 * c) * 16 + mlo;
            #pragma unroll
            for (int i = 0; i < 8; ++i) {
                int m = mr + i;
                float cc   = tanhf(acc[c][i]);
                float u    = sU[m * UST + n];
                float hold = bf2f(sH[m * NN + n]);
                float hn   = u * hold + (1.0f - u) * cc;
                if (MASK) { if (t >= sLen[m]) hn = hold; }   // TF length-freeze
                sH[m * NN + n] = f2bf(hn);
                if (WOUT) outp[(long)m * outStride + n] = hn;
            }
        }
    }
    __syncthreads();
}

// ---------------------------------------------------------------------------
// Persistent sequential kernel: one workgroup runs the full 1024-step scan.
// ---------------------------------------------------------------------------
__global__ __launch_bounds__(NTH, 1) void seq_kernel(
    const float* __restrict__ enc_in, const float* __restrict__ dec_in,
    const int*   __restrict__ lens,
    const float* __restrict__ enc_bg, const float* __restrict__ enc_bc,
    const float* __restrict__ dec_bg, const float* __restrict__ dec_bc,
    const bf16*  __restrict__ wpk, float* __restrict__ out)
{
    extern __shared__ unsigned char smem[];
    bf16*  sA   = (bf16*) (smem + SA_OFF);
    bf16*  sRH  = (bf16*) (smem + SRH_OFF);
    float* sU   = (float*)(smem + SU_OFF);
    bf16*  sH0  = (bf16*) (smem + SH0_OFF);
    bf16*  sH1  = (bf16*) (smem + SH1_OFF);
    int*   sLen = (int*)  (smem + SLEN_OFF);
    const int tid = threadIdx.x;

    for (int w = tid; w < NB * NN; w += NTH) { sH0[w] = f2bf(0.f); sH1[w] = f2bf(0.f); }
    if (tid < NB) sLen[tid] = lens[tid];
    __syncthreads();

    const bf16* eg0 = wpk;
    const bf16* ec0 = eg0 + GATE_ELEMS;
    const bf16* eg1 = wpk + (long)LAYER_ELEMS;
    const bf16* ec1 = eg1 + GATE_ELEMS;
    const bf16* dg0 = wpk + 2L * LAYER_ELEMS;
    const bf16* dc0 = dg0 + GATE_ELEMS;
    const bf16* dg1 = wpk + 3L * LAYER_ELEMS;
    const bf16* dc1 = dg1 + GATE_ELEMS;

    // ---- encoder: 512 steps, 2 layers, length-masked state freeze ----
    for (int t = 0; t < TE; ++t) {
        gru_layer<true,  true, false>(enc_in + (long)t * NN, (long)TE * NN, nullptr,
            sA, sRH, sU, sH0, sLen, t, eg0, ec0, enc_bg,       enc_bc,       nullptr, 0, tid);
        gru_layer<false, true, false>(nullptr, 0, sH0,
            sA, sRH, sU, sH1, sLen, t, eg1, ec1, enc_bg + 512, enc_bc + 256, nullptr, 0, tid);
    }
    // ---- decoder: 512 steps, teacher forcing, write h1 each step ----
    for (int t = 0; t < TD; ++t) {
        gru_layer<true,  false, false>(dec_in + (long)t * NN, (long)TD * NN, nullptr,
            sA, sRH, sU, sH0, sLen, t, dg0, dc0, dec_bg,       dec_bc,       nullptr, 0, tid);
        gru_layer<false, false, true >(nullptr, 0, sH0,
            sA, sRH, sU, sH1, sLen, t, dg1, dc1, dec_bg + 512, dec_bc + 256,
            out + (long)t * NN, (long)TD * NN, tid);
    }
}

// ---------------------------------------------------------------------------
extern "C" void kernel_launch(void* const* d_in, const int* in_sizes, int n_in,
                              void* d_out, int out_size, void* d_ws, size_t ws_size,
                              hipStream_t stream) {
    const float* enc_in  = (const float*)d_in[0];
    const float* dec_in  = (const float*)d_in[1];
    const int*   lens    = (const int*)  d_in[2];
    const float* enc_Wxg = (const float*)d_in[3];
    const float* enc_Whg = (const float*)d_in[4];
    const float* enc_bg  = (const float*)d_in[5];
    const float* enc_Wxc = (const float*)d_in[6];
    const float* enc_Whc = (const float*)d_in[7];
    const float* enc_bc  = (const float*)d_in[8];
    const float* dec_Wxg = (const float*)d_in[9];
    const float* dec_Whg = (const float*)d_in[10];
    const float* dec_bg  = (const float*)d_in[11];
    const float* dec_Wxc = (const float*)d_in[12];
    const float* dec_Whc = (const float*)d_in[13];
    const float* dec_bc  = (const float*)d_in[14];
    bf16*  wpk = (bf16*)d_ws;               // needs 4*LAYER_ELEMS*2 = 3 MiB
    float* out = (float*)d_out;

    const long total = 4L * LAYER_ELEMS;
    int blocks = (int)((total + 255) / 256);
    pack_weights_kernel<<<blocks, 256, 0, stream>>>(
        enc_Wxg, enc_Whg, enc_Wxc, enc_Whc,
        dec_Wxg, dec_Whg, dec_Wxc, dec_Whc, wpk);

    hipFuncSetAttribute((const void*)seq_kernel,
                        hipFuncAttributeMaxDynamicSharedMemorySize, SMEM_BYTES);
    seq_kernel<<<1, NTH, SMEM_BYTES, stream>>>(
        enc_in, dec_in, lens, enc_bg, enc_bc, dec_bg, dec_bc, wpk, out);
}
